// MHA_49057116455506
// MI455X (gfx1250) — matrix operand order repaired and checked by reading, not compiled, not run on hardware
//
#include <hip/hip_runtime.h>

// ---------------------------------------------------------------------------
// MHA block for MI455X (gfx1250): bf16 WMMA everywhere, fp32 softmax.
// B=2, S=2048, E=2048, H=16, D=128
// ---------------------------------------------------------------------------

typedef __bf16 bf16_t;
typedef __attribute__((ext_vector_type(16))) __bf16 v16bf;
typedef __attribute__((ext_vector_type(8)))  __bf16 v8bf;
typedef __attribute__((ext_vector_type(8)))  float  v8f;

static constexpr int Bb = 2;
static constexpr int Ss = 2048;
static constexpr int Ee = 2048;
static constexpr int Hh = 16;
static constexpr int Dd = 128;
static constexpr int E3 = 3 * Ee;            // 6144
static constexpr int Mtot = Bb * Ss;         // 4096

#define WMMA_BF16(a, b, c) \
  __builtin_amdgcn_wmma_f32_16x16x32_bf16(false, (a), false, (b), (short)0, (c), false, false)

// A-matrix 16x32 bf16 fragment (per ISA layout):
//   e in [0,8):  K = k0 + e + 8*half      -> 8 contiguous at base + 8*half
//   e in [8,16): K = k0 + e + 8 + 8*half  -> 8 contiguous at base + 16 + 8*half
__device__ __forceinline__ v16bf load_frag_a(const bf16_t* base, int half) {
  v8bf lo = *(const v8bf*)(base + 8 * half);
  v8bf hi = *(const v8bf*)(base + 16 + 8 * half);
  v16bf r;
#pragma unroll
  for (int i = 0; i < 8; ++i) { r[i] = lo[i]; r[8 + i] = hi[i]; }
  return r;
}

// B-matrix 32x16 bf16 fragment: lane holds column n = lane&15,
//   e in [0,8):  K = k0 + e + 8*half      -> 8 contiguous at base + 8*half
//   e in [8,16): K = k0 + e + 8 + 8*half  -> 8 contiguous at base + 16 + 8*half
__device__ __forceinline__ v16bf load_frag_b(const bf16_t* base, int half) {
  v8bf lo = *(const v8bf*)(base + 8 * half);
  v8bf hi = *(const v8bf*)(base + 16 + 8 * half);
  v16bf r;
#pragma unroll
  for (int i = 0; i < 8; ++i) { r[i] = lo[i]; r[8 + i] = hi[i]; }
  return r;
}

// ---------------------------------------------------------------------------
// fp32 -> bf16 conversion (4 elems/thread, all sizes divisible by 1024)
// ---------------------------------------------------------------------------
__global__ __launch_bounds__(256) void f2bf_kernel(const float* __restrict__ in,
                                                   bf16_t* __restrict__ out, int n4) {
  int i = (blockIdx.x * 256 + threadIdx.x);
  if (i < n4) {
    float4 v = ((const float4*)in)[i];
    v8bf* o = (v8bf*)out;  // unused; plain stores below
    bf16_t b0 = (bf16_t)v.x, b1 = (bf16_t)v.y, b2 = (bf16_t)v.z, b3 = (bf16_t)v.w;
    out[i * 4 + 0] = b0; out[i * 4 + 1] = b1;
    out[i * 4 + 2] = b2; out[i * 4 + 3] = b3;
    (void)o;
  }
}

// ---------------------------------------------------------------------------
// GEMM1: qkv[M,3E] = bf16( x[M,E] @ W_in[3E,E]^T + b_in ), plus transposed V
// One wave computes a 16(M) x 64(N) tile. 8 waves / block.
// ---------------------------------------------------------------------------
__global__ __launch_bounds__(256) void gemm_qkv_kernel(const bf16_t* __restrict__ xb,
                                                       const bf16_t* __restrict__ Wi,
                                                       const float* __restrict__ b_in,
                                                       bf16_t* __restrict__ qkv,
                                                       bf16_t* __restrict__ vT) {
  const int lane = threadIdx.x & 31;
  const int ncol = lane & 15;
  const int half = lane >> 4;
  const int wid = blockIdx.x * 8 + (threadIdx.x >> 5);
  const int TN = E3 / 64;            // 96 column-groups
  const int mt = wid / TN;
  const int nt0 = (wid % TN) * 4;    // 4 n-tiles of 16
  const int m0 = mt * 16;

  v8f acc[4] = {};
  const bf16_t* arow = xb + (size_t)(m0 + ncol) * Ee;
#pragma unroll 4
  for (int k0 = 0; k0 < Ee; k0 += 32) {
    __builtin_prefetch(arow + k0 + 256, 0, 0);
    v16bf af = load_frag_a(arow + k0, half);
#pragma unroll
    for (int j = 0; j < 4; ++j) {
      const bf16_t* brow = Wi + (size_t)((nt0 + j) * 16 + ncol) * Ee + k0;
      v16bf bfr = load_frag_b(brow, half);
      acc[j] = WMMA_BF16(af, bfr, acc[j]);
    }
  }

#pragma unroll
  for (int j = 0; j < 4; ++j) {
    const int n = (nt0 + j) * 16 + ncol;
    const float bias = b_in[n];
    v8bf pack;
#pragma unroll
    for (int r = 0; r < 8; ++r) {
      const int row = m0 + r + 8 * half;      // D layout: M = r + 8*half, N = lane&15
      bf16_t bv = (bf16_t)(acc[j][r] + bias);
      qkv[(size_t)row * E3 + n] = bv;
      pack[r] = bv;
    }
    if (n >= 2 * Ee) {                        // V range: also store transposed copy
      const int batch = m0 >> 11;             // tile never crosses the S=2048 boundary
      const int dglob = n - 2 * Ee;
      const int hh = dglob >> 7;
      const int dd = dglob & 127;
      const size_t vtoff =
          ((size_t)((batch * Hh + hh)) * Dd + dd) * Ss + (m0 & (Ss - 1)) + 8 * half;
      *(v8bf*)(vT + vtoff) = pack;            // 8 consecutive s positions
    }
  }
}

// ---------------------------------------------------------------------------
// Flash attention: 1 wave = 16 query rows, online softmax over S in 32-col steps
// block = 128 threads (4 waves) -> 64 queries; grid = B*H*(S/64)
// ---------------------------------------------------------------------------
__global__ __launch_bounds__(128) void attn_kernel(const bf16_t* __restrict__ qkv,
                                                   const bf16_t* __restrict__ vT,
                                                   const unsigned char* __restrict__ mask,
                                                   bf16_t* __restrict__ ctx) {
  __shared__ __align__(16) bf16_t pbuf[4][16 * 32];
  const int lane = threadIdx.x & 31;
  const int wid = threadIdx.x >> 5;
  const int ncol = lane & 15;
  const int half = lane >> 4;
  const int qblocks = Ss / 64;  // 32
  const int bh = blockIdx.x / qblocks;
  const int l0 = (blockIdx.x % qblocks) * 64 + wid * 16;
  const int b = bh / Hh;
  const int h = bh % Hh;

  const bf16_t* qbase = qkv + (size_t)(b * Ss) * E3 + h * Dd;
  const bf16_t* kbase = qkv + (size_t)(b * Ss) * E3 + Ee + h * Dd;
  const bf16_t* vtb = vT + (size_t)bh * Dd * Ss;
  const unsigned char* mrow = mask + b * Ss;
  bf16_t* myp = pbuf[wid];

  // Q fragments: row = l0 + (lane&15), 4 K-steps of 32 cover D=128
  v16bf qf[4];
  {
    const bf16_t* qrow = qbase + (size_t)(l0 + ncol) * E3;
#pragma unroll
    for (int i = 0; i < 4; ++i) qf[i] = load_frag_a(qrow + i * 32, half);
  }

  v8f cacc[8] = {};
  float mstate[8], lstate[8];
#pragma unroll
  for (int r = 0; r < 8; ++r) { mstate[r] = -__builtin_inff(); lstate[r] = 0.0f; }

  const float scale = 0.08838834764831845f;  // 1/sqrt(128)

  for (int s0 = 0; s0 < Ss; s0 += 32) {
    // ---- scores: two 16x16 tiles (cols s0..s0+15, s0+16..s0+31) ----
    v8f sa = {}, sb = {};
    const bf16_t* krow0 = kbase + (size_t)(s0 + ncol) * E3;
    const bf16_t* krow1 = kbase + (size_t)(s0 + 16 + ncol) * E3;
#pragma unroll
    for (int i = 0; i < 4; ++i) {
      v16bf kb0 = load_frag_b(krow0 + i * 32, half);
      sa = WMMA_BF16(qf[i], kb0, sa);
      v16bf kb1 = load_frag_b(krow1 + i * 32, half);
      sb = WMMA_BF16(qf[i], kb1, sb);
    }

    const bool mk0 = mrow[s0 + ncol] != 0;
    const bool mk1 = mrow[s0 + 16 + ncol] != 0;

    float tmax[8];
#pragma unroll
    for (int r = 0; r < 8; ++r) {
      float x0 = mk0 ? -__builtin_inff() : sa[r] * scale;
      float x1 = mk1 ? -__builtin_inff() : sb[r] * scale;
      sa[r] = x0; sb[r] = x1;
      tmax[r] = fmaxf(x0, x1);
    }
    // row max across the 16-lane half-group (rows live across lanes in C/D layout)
    for (int mxs = 1; mxs < 16; mxs <<= 1) {
#pragma unroll
      for (int r = 0; r < 8; ++r) tmax[r] = fmaxf(tmax[r], __shfl_xor(tmax[r], mxs, 32));
    }

    float pr0[8], pr1[8], scl[8], rsum[8];
#pragma unroll
    for (int r = 0; r < 8; ++r) {
      float mold = mstate[r];
      float mnew = fmaxf(mold, tmax[r]);
      float f = (mold == mnew) ? 1.0f : __expf(mold - mnew);
      mstate[r] = mnew;
      pr0[r] = __expf(sa[r] - mnew);
      pr1[r] = __expf(sb[r] - mnew);
      scl[r] = f;
      rsum[r] = pr0[r] + pr1[r];
    }
    for (int mxs = 1; mxs < 16; mxs <<= 1) {
#pragma unroll
      for (int r = 0; r < 8; ++r) rsum[r] += __shfl_xor(rsum[r], mxs, 32);
    }
#pragma unroll
    for (int r = 0; r < 8; ++r) lstate[r] = lstate[r] * scl[r] + rsum[r];
#pragma unroll
    for (int nt = 0; nt < 8; ++nt)
#pragma unroll
      for (int r = 0; r < 8; ++r) cacc[nt][r] *= scl[r];

    // ---- transpose P through LDS: D-layout (row across lanes) -> A-layout ----
#pragma unroll
    for (int r = 0; r < 8; ++r) {
      myp[(r + 8 * half) * 32 + ncol] = (bf16_t)pr0[r];
      myp[(r + 8 * half) * 32 + 16 + ncol] = (bf16_t)pr1[r];
    }
    asm volatile("s_wait_dscnt 0" ::: "memory");
    v16bf pa = load_frag_a(myp + ncol * 32, half);

    // ---- ctx += P @ V  (B frags contiguous from transposed V) ----
#pragma unroll
    for (int nt = 0; nt < 8; ++nt) {
      v16bf vb = load_frag_b(vtb + (size_t)(nt * 16 + ncol) * Ss + s0, half);
      cacc[nt] = WMMA_BF16(pa, vb, cacc[nt]);
    }
  }

  // ---- epilogue: ctx /= l, store bf16 [B*S, E] ----
#pragma unroll
  for (int r = 0; r < 8; ++r) {
    const float inv = 1.0f / lstate[r];
    const int row = l0 + r + 8 * half;
    bf16_t* crow = ctx + (size_t)(b * Ss + row) * Ee + h * Dd;
#pragma unroll
    for (int nt = 0; nt < 8; ++nt) crow[nt * 16 + ncol] = (bf16_t)(cacc[nt][r] * inv);
  }
}

// ---------------------------------------------------------------------------
// GEMM3: out[M,E] = ctx[M,E] @ W_out[E,E]^T + b_out  (fp32 output)
// ---------------------------------------------------------------------------
__global__ __launch_bounds__(256) void gemm_out_kernel(const bf16_t* __restrict__ ctxb,
                                                       const bf16_t* __restrict__ Wo,
                                                       const float* __restrict__ b_out,
                                                       float* __restrict__ out) {
  const int lane = threadIdx.x & 31;
  const int ncol = lane & 15;
  const int half = lane >> 4;
  const int wid = blockIdx.x * 8 + (threadIdx.x >> 5);
  const int TN = Ee / 64;            // 32 column-groups
  const int mt = wid / TN;
  const int nt0 = (wid % TN) * 4;
  const int m0 = mt * 16;

  v8f acc[4] = {};
  const bf16_t* arow = ctxb + (size_t)(m0 + ncol) * Ee;
#pragma unroll 4
  for (int k0 = 0; k0 < Ee; k0 += 32) {
    __builtin_prefetch(arow + k0 + 256, 0, 0);
    v16bf af = load_frag_a(arow + k0, half);
#pragma unroll
    for (int j = 0; j < 4; ++j) {
      const bf16_t* brow = Wo + (size_t)((nt0 + j) * 16 + ncol) * Ee + k0;
      v16bf bfr = load_frag_b(brow, half);
      acc[j] = WMMA_BF16(af, bfr, acc[j]);
    }
  }
#pragma unroll
  for (int j = 0; j < 4; ++j) {
    const int n = (nt0 + j) * 16 + ncol;
    const float bias = b_out[n];
#pragma unroll
    for (int r = 0; r < 8; ++r) {
      const int row = m0 + r + 8 * half;
      out[(size_t)row * Ee + n] = acc[j][r] + bias;
    }
  }
}

// ---------------------------------------------------------------------------
// Launch
// ---------------------------------------------------------------------------
extern "C" void kernel_launch(void* const* d_in, const int* in_sizes, int n_in,
                              void* d_out, int out_size, void* d_ws, size_t ws_size,
                              hipStream_t stream) {
  const float* x = (const float*)d_in[0];       // [B,S,E]
  const float* W_in = (const float*)d_in[1];    // [3E,E]
  const float* b_in = (const float*)d_in[2];    // [3E]
  const float* W_out = (const float*)d_in[3];   // [E,E]
  const float* b_out = (const float*)d_in[4];   // [E]
  const unsigned char* mask = (const unsigned char*)d_in[5];  // [B,S] bool
  float* out = (float*)d_out;                   // [B,S,E]

  char* ws = (char*)d_ws;
  size_t off = 0;
  bf16_t* xb   = (bf16_t*)(ws + off); off += (size_t)Mtot * Ee * 2;        // 16 MB
  bf16_t* Wib  = (bf16_t*)(ws + off); off += (size_t)E3 * Ee * 2;          // 24 MB
  bf16_t* Wob  = (bf16_t*)(ws + off); off += (size_t)Ee * Ee * 2;          // 8 MB
  bf16_t* qkvb = (bf16_t*)(ws + off); off += (size_t)Mtot * E3 * 2;        // 48 MB
  bf16_t* vTb  = (bf16_t*)(ws + off); off += (size_t)Bb * Hh * Dd * Ss * 2; // 16 MB
  bf16_t* ctxb = (bf16_t*)(ws + off); off += (size_t)Mtot * Ee * 2;        // 16 MB
  (void)ws_size;

  // 1) fp32 -> bf16 conversions
  {
    int n4;
    n4 = (Mtot * Ee) / 4;
    f2bf_kernel<<<(n4 + 255) / 256, 256, 0, stream>>>(x, xb, n4);
    n4 = (E3 * Ee) / 4;
    f2bf_kernel<<<(n4 + 255) / 256, 256, 0, stream>>>(W_in, Wib, n4);
    n4 = (Ee * Ee) / 4;
    f2bf_kernel<<<(n4 + 255) / 256, 256, 0, stream>>>(W_out, Wob, n4);
  }

  // 2) QKV projection (+ transposed V side-store)
  {
    int tiles = (Mtot / 16) * (E3 / 64);  // 24576 wave-tiles
    gemm_qkv_kernel<<<tiles / 8, 256, 0, stream>>>(xb, Wib, b_in, qkvb, vTb);
  }

  // 3) masked flash attention
  {
    int blocks = Bb * Hh * (Ss / 64);  // 1024
    attn_kernel<<<blocks, 128, 0, stream>>>(qkvb, vTb, mask, ctxb);
  }

  // 4) output projection (fp32 out)
  {
    int tiles = (Mtot / 16) * (Ee / 64);  // 8192 wave-tiles
    gemm_out_kernel<<<tiles / 8, 256, 0, stream>>>(ctxb, Wob, b_out, out);
  }
}
